// TransliterationModel_35399120454108
// MI455X (gfx1250) — matrix operand order repaired and checked by reading, not compiled, not run on hardware
//
#include <hip/hip_runtime.h>

// ---------------------------------------------------------------------------
// Problem constants (match reference)
// ---------------------------------------------------------------------------
#define BB   256   // batch
#define SS   128   // src len
#define TT   128   // tgt len
#define EE   512   // embed dim
#define HH   1024  // hidden (decoder / concat)
#define HD   512   // per-direction hidden
#define VV   256   // vocab

typedef __attribute__((ext_vector_type(16))) __bf16 bf16x16;
typedef __attribute__((ext_vector_type(8)))  float  floatx8;
typedef __attribute__((ext_vector_type(4)))  int    intx4;

#define AS1 __attribute__((address_space(1)))
#define AS3 __attribute__((address_space(3)))

#if defined(__has_builtin)
# if __has_builtin(__builtin_amdgcn_global_load_async_to_lds_b128)
#  define USE_ASYNC_LDS 1
# endif
#endif

// 16B global -> LDS copy, async (GLOBAL_LOAD_ASYNC_TO_LDS_B128) when available.
// ROCm signature: (global int4* src, local int4* dst, imm offset, imm cpol).
__device__ __forceinline__ void cp16_to_lds(void* lds, const void* glob)
{
#ifdef USE_ASYNC_LDS
    // LDS flat addresses carry the LDS offset in the low 32 bits (aperture scheme).
    __builtin_amdgcn_global_load_async_to_lds_b128(
        (AS1 intx4*)(uintptr_t)glob,
        (AS3 intx4*)(unsigned)(uintptr_t)lds, 0, 0);
#else
    *(uint4*)lds = *(const uint4*)glob;
#endif
}

__device__ __forceinline__ void wait_async_lds()
{
#ifdef USE_ASYNC_LDS
# if __has_builtin(__builtin_amdgcn_s_wait_asynccnt)
    __builtin_amdgcn_s_wait_asynccnt(0);
# else
    asm volatile("s_wait_asynccnt 0x0" ::: "memory");
# endif
#endif
}

// ---------------------------------------------------------------------------
// Dual-source WMMA GEMM: C[M,N] = A1@W1^T + A2@W2^T + bias.
// A bf16 row-major. W pre-packed in tile order:
//   Wp[nb][kb][k][n] (64x32 tile = contiguous 4KB, element [k*64+n] = W[nb*64+n][kb*32+k])
// 256 threads = 8 waves; block tile 128(M) x 64(N), K-step 32; wave w owns
// rows [16w,16w+16) x 64 N (4 wmma accumulators).
// DOUBLE-BUFFERED: async b128 global->LDS copies for tile k+1 are issued
// right after the barrier, overlapping the ds_load_b128 + v_wmma of tile k.
// ---------------------------------------------------------------------------
__global__ __launch_bounds__(256) void gemm2_bf16(
    const __bf16* __restrict__ A1, int lda1, const __bf16* __restrict__ W1p, int K1,
    const __bf16* __restrict__ A2, int lda2, const __bf16* __restrict__ W2p, int K2,
    float* __restrict__ C, __bf16* __restrict__ Cbf, int ldc,
    const float* __restrict__ bias)
{
    __shared__ __align__(16) __bf16 As[2][128][32];
    __shared__ __align__(16) __bf16 Bs[2][32][72];   // [K][N], padded row

    const int  tid  = threadIdx.x;
    const int  lane = tid & 31;
    const int  wid  = tid >> 5;
    const long M0   = (long)blockIdx.x * 128;
    const int  nb   = blockIdx.y;                    // 64-wide N block

    const int nst1 = K1 >> 5;
    const int nst2 = (A2 != nullptr) ? (K2 >> 5) : 0;
    const int nst  = nst1 + nst2;

    // issue async staging of K-step `st` into buffer `buf`
    auto issue = [&](int st, int buf) {
        const __bf16* A; const __bf16* Wp; int lda, kb, nk;
        if (st < nst1) { A = A1; Wp = W1p; lda = lda1; kb = st;        nk = nst1; }
        else           { A = A2; Wp = W2p; lda = lda2; kb = st - nst1; nk = nst2; }
        const int k0 = kb * 32;
        // A tile 128x32: 512 x 16B chunks, 2 per thread
#pragma unroll
        for (int i = 0; i < 2; ++i) {
            int chunk = tid * 2 + i;
            int r  = chunk >> 2;
            int cc = (chunk & 3) * 8;
            cp16_to_lds(&As[buf][r][cc], A + (M0 + r) * (long)lda + k0 + cc);
        }
        // W tile: contiguous 4KB packed block, 1 x 16B chunk per thread
        cp16_to_lds(&Bs[buf][tid >> 3][(tid & 7) * 8],
                    Wp + ((long)nb * nk + kb) * 2048 + tid * 8);
    };

    const floatx8 vzero = {0.f,0.f,0.f,0.f,0.f,0.f,0.f,0.f};
    floatx8 acc[4];
#pragma unroll
    for (int i = 0; i < 4; ++i) acc[i] = vzero;

    issue(0, 0);
    for (int st = 0; st < nst; ++st) {
        const int buf = st & 1;
        wait_async_lds();            // this wave's staging for `st` complete
        __syncthreads();             // everyone's staging visible
        if (st + 1 < nst) issue(st + 1, buf ^ 1);   // overlap with compute

        // A fragment (ISA 7.12.2 16-bit A 16x32): 2 x ds_load_b128
        union { uint4 u[2]; bf16x16 v; } afu;
        {
            int m  = lane & 15;
            int kb = (lane >> 4) * 8;
            const __bf16* ar = &As[buf][wid * 16 + m][0];
            afu.u[0] = *(const uint4*)(ar + kb);
            afu.u[1] = *(const uint4*)(ar + 16 + kb);
        }
        // 4 x (B fragment + WMMA); lane = N column, 16 K values in table order
#pragma unroll
        for (int nt = 0; nt < 4; ++nt) {
            union { uint4 u[2]; bf16x16 v; } bfu;
            const int bn = nt * 16 + (lane & 15);
            const int bk = (lane >> 4) * 8;
#pragma unroll
            for (int s = 0; s < 8; ++s) {
                bfu.v[s]     = Bs[buf][bk + s][bn];
                bfu.v[8 + s] = Bs[buf][16 + bk + s][bn];
            }
            acc[nt] = __builtin_amdgcn_wmma_f32_16x16x32_bf16(
                false, afu.v, false, bfu.v, (short)0, acc[nt], false, false);
        }
        // no trailing barrier needed: a wave's ds reads of buf retire before
        // its wmma, which precedes its next barrier arrival; buf is refilled
        // only after that next barrier.
    }

    // ---- store D (n = lane&15, m = (lane>>4)*8 + j)
    const int nl = lane & 15;
    const int mb = (lane >> 4) * 8;
#pragma unroll
    for (int nt = 0; nt < 4; ++nt) {
        int   n  = nb * 64 + nt * 16 + nl;
        float bv = bias ? bias[n] : 0.f;
#pragma unroll
        for (int j = 0; j < 8; ++j) {
            long m = M0 + wid * 16 + mb + j;
            float v = acc[nt][j] + bv;
            if (Cbf) Cbf[m * (long)ldc + n] = (__bf16)v;
            else     C  [m * (long)ldc + n] = v;
        }
    }
}

// ---------------------------------------------------------------------------
// LSTM cell elementwise: gates g (B x grs) chunks [i|f|g|o] of width hid.
// Updates c (f32) and h (bf16) at column colOff, row stride srs; optional
// write of h into a sequence buffer at seqBase.
// ---------------------------------------------------------------------------
__global__ void lstm_cell(const float* __restrict__ g, int grs, int hid,
                          float* __restrict__ c, __bf16* __restrict__ hbf,
                          int colOff, int srs,
                          __bf16* __restrict__ seq, long seqBase, int Bn)
{
    long idx = (long)blockIdx.x * blockDim.x + threadIdx.x;
    if (idx >= (long)Bn * hid) return;
    int b = (int)(idx / hid), j = (int)(idx % hid);
    const float* gr = g + (long)b * grs;
    float gi = gr[j], gf = gr[hid + j], gg = gr[2 * hid + j], go = gr[3 * hid + j];
    long  si = (long)b * srs + colOff + j;
    float cv = c[si];
    float is = 1.f / (1.f + __expf(-gi));
    float fs = 1.f / (1.f + __expf(-gf));
    float os = 1.f / (1.f + __expf(-go));
    cv = fs * cv + is * tanhf(gg);
    float hv = os * tanhf(cv);
    c[si] = cv;
    __bf16 hb = (__bf16)hv;
    hbf[si] = hb;
    if (seq) seq[seqBase + si] = hb;
}

// ---------------------------------------------------------------------------
// Attention step; one 256-thread block per batch element.
// enc_proj/enc_out bf16: together 134 MB -> resident in MI455X 192 MB L2
// across all 127 decoder steps.
// ---------------------------------------------------------------------------
__global__ __launch_bounds__(256) void attn_step(
    const float*  __restrict__ q,         // (B,H) f32
    const __bf16* __restrict__ enc_proj,  // (S,B,H) bf16
    const __bf16* __restrict__ enc_out,   // (S,B,H) bf16
    const float*  __restrict__ w2, const float* __restrict__ b2,
    const int*    __restrict__ src,       // (B,S)
    const int*    __restrict__ tgt,       // (B,T)
    const float*  __restrict__ dec_embed, // (V,E) f32
    __bf16*       __restrict__ rin,       // (B, E+H) bf16
    float*        __restrict__ attn_out,  // (B, T-1, S) region of d_out
    int t)
{
    __shared__ float qs[HH];
    __shared__ float w2s[HH];
    __shared__ float en[SS];

    const int b = blockIdx.x;
    const int tid = threadIdx.x;
    *(float4*)&qs[tid * 4]  = *(const float4*)&q[(long)b * HH + tid * 4];
    *(float4*)&w2s[tid * 4] = *(const float4*)&w2[tid * 4];
    __syncthreads();

    const int lane = tid & 31, wid = tid >> 5;
    for (int s = wid; s < SS; s += 8) {
        const __bf16* ep = enc_proj + ((long)s * BB + b) * HH;
        float sum = 0.f;
#pragma unroll
        for (int i = 0; i < 4; ++i) {
            int h = (lane + 32 * i) * 8;          // 8 contiguous bf16 per lane
            union { uint4 u; __bf16 e[8]; } blk;
            blk.u = *(const uint4*)(ep + h);
#pragma unroll
            for (int j = 0; j < 8; ++j)
                sum += tanhf((float)blk.e[j] + qs[h + j]) * w2s[h + j];
        }
        for (int off = 16; off > 0; off >>= 1)
            sum += __shfl_down(sum, off, 32);
        if (lane == 0)
            en[s] = (src[b * SS + s] != 0) ? (sum + b2[0]) : -__builtin_inff();
    }
    __syncthreads();

    if (tid == 0) {
        float mx = -__builtin_inff();
        for (int s = 0; s < SS; ++s) mx = fmaxf(mx, en[s]);
        float sum = 0.f;
        for (int s = 0; s < SS; ++s) { float e = __expf(en[s] - mx); en[s] = e; sum += e; }
        float inv = 1.f / sum;
        for (int s = 0; s < SS; ++s) en[s] *= inv;
    }
    __syncthreads();

    if (tid < SS)
        attn_out[((long)b * (TT - 1) + t) * SS + tid] = en[tid];

    // ctx: 4 consecutive h per thread, 8B vector loads/stores
    const int h0 = tid * 4;
    float acc4[4] = {0.f, 0.f, 0.f, 0.f};
    for (int s = 0; s < SS; ++s) {
        float a = en[s];
        union { uint2 u; __bf16 e[4]; } blk;
        blk.u = *(const uint2*)(enc_out + ((long)s * BB + b) * HH + h0);
#pragma unroll
        for (int i = 0; i < 4; ++i) acc4[i] += a * (float)blk.e[i];
    }
    {
        union { uint2 u; __bf16 e[4]; } ob;
#pragma unroll
        for (int i = 0; i < 4; ++i) ob.e[i] = (__bf16)acc4[i];
        *(uint2*)(rin + (long)b * (EE + HH) + EE + h0) = ob.u;
    }
    // embedding half of rin
    if (tid < 128) {
        int tok = tgt[b * TT + t];
        const float* de = dec_embed + (long)tok * EE + tid * 4;
        union { uint2 u; __bf16 e[4]; } eb;
#pragma unroll
        for (int i = 0; i < 4; ++i) eb.e[i] = (__bf16)de[i];
        *(uint2*)(rin + (long)b * (EE + HH) + tid * 4) = eb.u;
    }
}

// ---------------------------------------------------------------------------
// f32 weight -> bf16 in WMMA tile order:
//   d[((n/64)*(K/32) + k/32)*2048 + (k%32)*64 + (n%64)] = s[n*srs + coff + k]
// 4 consecutive packed outputs per thread (same k, n..n+3).
// ---------------------------------------------------------------------------
__global__ void pack_w(const float* __restrict__ s, int srs, int coff,
                       __bf16* __restrict__ d, int K, long n4)
{
    long i = (long)blockIdx.x * blockDim.x + threadIdx.x;
    if (i >= n4) return;
    long o    = i * 4;
    int  nk   = K >> 5;
    long tile = o >> 11;
    int  nb   = (int)(tile / nk);
    int  kb   = (int)(tile % nk);
    int  k    = kb * 32 + (int)((o >> 6) & 31);
    int  n    = nb * 64 + (int)(o & 63);
    union { uint2 u; __bf16 e[4]; } ob;
#pragma unroll
    for (int j = 0; j < 4; ++j)
        ob.e[j] = (__bf16)s[(long)(n + j) * srs + coff + k];
    *(uint2*)(d + o) = ob.u;
}

// Encoder embedding gather: x[(t*B+b)*E + e..e+3] = enc_embed[src[b,t], e..e+3]
__global__ void gather_embed(const int* __restrict__ src, const float* __restrict__ emb,
                             __bf16* __restrict__ x, long n4)
{
    long i = (long)blockIdx.x * blockDim.x + threadIdx.x;
    if (i >= n4) return;
    int  e = (int)(i % (EE / 4)) * 4;
    long r = i / (EE / 4);
    int  b = (int)(r % BB);
    int  t = (int)(r / BB);
    const float* sp = emb + (long)src[b * SS + t] * EE + e;
    union { uint2 u; __bf16 v[4]; } ob;
#pragma unroll
    for (int j = 0; j < 4; ++j) ob.v[j] = (__bf16)sp[j];
    *(uint2*)(x + r * EE + e) = ob.u;
}

// outputs[:,0,:] = 0
__global__ void zero_t0(float* __restrict__ out)
{
    int i = blockIdx.x * 256 + threadIdx.x;     // < B*V
    out[(long)(i / VV) * (TT * VV) + (i % VV)] = 0.f;
}

// ---------------------------------------------------------------------------
// Host orchestration
// ---------------------------------------------------------------------------
extern "C" void kernel_launch(void* const* d_in, const int* in_sizes, int n_in,
                              void* d_out, int out_size, void* d_ws, size_t ws_size,
                              hipStream_t stream)
{
    const int*   src       = (const int*)  d_in[0];
    const int*   tgt       = (const int*)  d_in[1];
    const float* enc_embed = (const float*)d_in[2];
    const float* dec_embed = (const float*)d_in[3];
    const float* l0f_wih = (const float*)d_in[4],  *l0f_whh = (const float*)d_in[5],  *l0f_b = (const float*)d_in[6];
    const float* l0b_wih = (const float*)d_in[7],  *l0b_whh = (const float*)d_in[8],  *l0b_b = (const float*)d_in[9];
    const float* l1f_wih = (const float*)d_in[10], *l1f_whh = (const float*)d_in[11], *l1f_b = (const float*)d_in[12];
    const float* l1b_wih = (const float*)d_in[13], *l1b_whh = (const float*)d_in[14], *l1b_b = (const float*)d_in[15];
    const float* d0_wih  = (const float*)d_in[16], *d0_whh  = (const float*)d_in[17], *d0_b  = (const float*)d_in[18];
    const float* d1_wih  = (const float*)d_in[19], *d1_whh  = (const float*)d_in[20], *d1_b  = (const float*)d_in[21];
    const float* attn_w1 = (const float*)d_in[22], *attn_b1 = (const float*)d_in[23];
    const float* attn_w2 = (const float*)d_in[24], *attn_b2 = (const float*)d_in[25];
    const float* fc_w    = (const float*)d_in[26], *fc_b    = (const float*)d_in[27];

    float* out      = (float*)d_out;
    float* attn_out = out + (long)BB * TT * VV;

    // ---- workspace carve-out (~310 MB total)
    char* p = (char*)d_ws;
    auto alloc = [&](size_t bytes) -> char* {
        char* r = p; p += (bytes + 255) & ~(size_t)255; return r;
    };
    __bf16* xbuf     = (__bf16*)alloc((size_t)SS * BB * EE * 2);   // 32 MB
    __bf16* x1       = (__bf16*)alloc((size_t)SS * BB * HH * 2);   // 64 MB
    __bf16* enc_out  = (__bf16*)alloc((size_t)SS * BB * HH * 2);   // 64 MB
    __bf16* enc_proj = (__bf16*)alloc((size_t)SS * BB * HH * 2);   // 64 MB (L2-resident w/ enc_out)
    float*  gbuf     = (float*) alloc((size_t)BB * 4 * HH * 4);    // 4 MB (gates)
    float*  qbuf     = (float*) alloc((size_t)BB * HH * 4);
    __bf16* rin      = (__bf16*)alloc((size_t)BB * (EE + HH) * 2);
    __bf16* hbf0     = (__bf16*)alloc((size_t)BB * HH * 2);
    __bf16* hbf1     = (__bf16*)alloc((size_t)BB * HH * 2);
    float*  cS0      = (float*) alloc((size_t)BB * HH * 4);
    float*  cS1      = (float*) alloc((size_t)BB * HH * 4);
    auto walloc = [&](size_t elems) -> __bf16* { return (__bf16*)alloc(elems * 2); };
    __bf16 *wl0f_ih = walloc(2048*512),  *wl0f_hh = walloc(2048*512);
    __bf16 *wl0b_ih = walloc(2048*512),  *wl0b_hh = walloc(2048*512);
    __bf16 *wl1f_ih = walloc(2048*1024), *wl1f_hh = walloc(2048*512);
    __bf16 *wl1b_ih = walloc(2048*1024), *wl1b_hh = walloc(2048*512);
    __bf16 *wd0_ih  = walloc(4096*1536), *wd0_hh  = walloc(4096*1024);
    __bf16 *wd1_ih  = walloc(4096*1024), *wd1_hh  = walloc(4096*1024);
    __bf16 *w1enc   = walloc(1024*1024), *w1q     = walloc(1024*1024);
    __bf16 *fcw     = walloc(256*1024);
    (void)ws_size; (void)in_sizes; (void)n_in; (void)out_size;

    // pack weights into WMMA tile order (bf16)
    auto pack = [&](const float* s, int srs, int coff, __bf16* d, int K, long N) {
        long n4 = N * (long)K / 4;
        pack_w<<<dim3((unsigned)((n4 + 255) / 256)), 256, 0, stream>>>(s, srs, coff, d, K, n4);
    };
    pack(l0f_wih, 512, 0, wl0f_ih, 512, 2048);  pack(l0f_whh, 512, 0, wl0f_hh, 512, 2048);
    pack(l0b_wih, 512, 0, wl0b_ih, 512, 2048);  pack(l0b_whh, 512, 0, wl0b_hh, 512, 2048);
    pack(l1f_wih, 1024,0, wl1f_ih, 1024,2048);  pack(l1f_whh, 512, 0, wl1f_hh, 512, 2048);
    pack(l1b_wih, 1024,0, wl1b_ih, 1024,2048);  pack(l1b_whh, 512, 0, wl1b_hh, 512, 2048);
    pack(d0_wih, 1536, 0, wd0_ih, 1536, 4096);  pack(d0_whh, 1024, 0, wd0_hh, 1024, 4096);
    pack(d1_wih, 1024, 0, wd1_ih, 1024, 4096);  pack(d1_whh, 1024, 0, wd1_hh, 1024, 4096);
    pack(attn_w1, 2048, 0,    w1enc, 1024, 1024);
    pack(attn_w1, 2048, 1024, w1q,   1024, 1024);
    pack(fc_w, 1024, 0, fcw, 1024, 256);

    // zero initial states
    (void)hipMemsetAsync(hbf0, 0, (size_t)BB * HH * 2, stream);
    (void)hipMemsetAsync(hbf1, 0, (size_t)BB * HH * 2, stream);
    (void)hipMemsetAsync(cS0,  0, (size_t)BB * HH * 4, stream);
    (void)hipMemsetAsync(cS1,  0, (size_t)BB * HH * 4, stream);
    zero_t0<<<dim3((BB * VV) / 256), 256, 0, stream>>>(out);

    // encoder input embedding gather (bf16, layout (S,B,E))
    {
        long n4 = (long)SS * BB * EE / 4;
        gather_embed<<<dim3((unsigned)(n4 / 256)), 256, 0, stream>>>(src, enc_embed, xbuf, n4);
    }

    const dim3 gStep(2, 32);          // M=256/128, N=2048/64
    float* gbufB = gbuf + (size_t)BB * 2048;

    // ---- encoder layer 0 (bidirectional): fused [x_t|h] @ [Wih;Whh]^T + b
    for (int s = 0; s < SS; ++s) {
        int tb = SS - 1 - s;
        gemm2_bf16<<<gStep, 256, 0, stream>>>(
            xbuf + (size_t)s  * BB * EE, EE, wl0f_ih, EE,
            hbf0,       HH, wl0f_hh, HD, gbuf,  nullptr, 2048, l0f_b);
        gemm2_bf16<<<gStep, 256, 0, stream>>>(
            xbuf + (size_t)tb * BB * EE, EE, wl0b_ih, EE,
            hbf0 + HD,  HH, wl0b_hh, HD, gbufB, nullptr, 2048, l0b_b);
        lstm_cell<<<dim3((BB * HD) / 256), 256, 0, stream>>>(
            gbuf,  2048, HD, cS0, hbf0, 0,  HH, x1, (long)s  * BB * HH, BB);
        lstm_cell<<<dim3((BB * HD) / 256), 256, 0, stream>>>(
            gbufB, 2048, HD, cS0, hbf0, HD, HH, x1, (long)tb * BB * HH, BB);
    }

    // ---- encoder layer 1 (bidirectional), input x1 (S,B,1024)
    for (int s = 0; s < SS; ++s) {
        int tb = SS - 1 - s;
        gemm2_bf16<<<gStep, 256, 0, stream>>>(
            x1 + (size_t)s  * BB * HH, HH, wl1f_ih, HH,
            hbf1,      HH, wl1f_hh, HD, gbuf,  nullptr, 2048, l1f_b);
        gemm2_bf16<<<gStep, 256, 0, stream>>>(
            x1 + (size_t)tb * BB * HH, HH, wl1b_ih, HH,
            hbf1 + HD, HH, wl1b_hh, HD, gbufB, nullptr, 2048, l1b_b);
        lstm_cell<<<dim3((BB * HD) / 256), 256, 0, stream>>>(
            gbuf,  2048, HD, cS1, hbf1, 0,  HH, enc_out, (long)s  * BB * HH, BB);
        lstm_cell<<<dim3((BB * HD) / 256), 256, 0, stream>>>(
            gbufB, 2048, HD, cS1, hbf1, HD, HH, enc_out, (long)tb * BB * HH, BB);
    }

    // ---- enc_proj = enc_out @ w1_enc^T  (M=32768, N=1024, K=1024) -> bf16
    gemm2_bf16<<<dim3(256, 16), 256, 0, stream>>>(
        enc_out, HH, w1enc, HH, nullptr, 0, nullptr, 0,
        nullptr, enc_proj, HH, nullptr);

    // ---- decoder: 127 steps; states hbf0/cS0, hbf1/cS1 start as encoder finals
    for (int t = 0; t < TT - 1; ++t) {
        gemm2_bf16<<<dim3(2, 16), 256, 0, stream>>>(
            hbf1, HH, w1q, HH, nullptr, 0, nullptr, 0,
            qbuf, nullptr, HH, attn_b1);
        attn_step<<<dim3(BB), 256, 0, stream>>>(
            qbuf, enc_proj, enc_out, attn_w2, attn_b2,
            src, tgt, dec_embed, rin, attn_out, t);
        gemm2_bf16<<<dim3(2, 64), 256, 0, stream>>>(
            rin, EE + HH, wd0_ih, EE + HH,
            hbf0, HH, wd0_hh, HH, gbuf, nullptr, 4096, d0_b);
        lstm_cell<<<dim3((BB * HH) / 256), 256, 0, stream>>>(
            gbuf, 4096, HH, cS0, hbf0, 0, HH, nullptr, 0, BB);
        gemm2_bf16<<<dim3(2, 64), 256, 0, stream>>>(
            hbf0, HH, wd1_ih, HH,
            hbf1, HH, wd1_hh, HH, gbuf, nullptr, 4096, d1_b);
        lstm_cell<<<dim3((BB * HH) / 256), 256, 0, stream>>>(
            gbuf, 4096, HH, cS1, hbf1, 0, HH, nullptr, 0, BB);
        gemm2_bf16<<<dim3(2, 4), 256, 0, stream>>>(
            hbf1, HH, fcw, HH, nullptr, 0, nullptr, 0,
            out + (size_t)(t + 1) * VV, nullptr, TT * VV, fc_b);
    }
}
